// LSTMSimple_25786983645490
// MI455X (gfx1250) — matrix operand order repaired and checked by reading, not compiled, not run on hardware
//
#include <hip/hip_runtime.h>
#include <hip/hip_bf16.h>

// Problem constants (from reference): B=64, T=512, F=64, H=512
#define BSZ 64
#define TT  512
#define FF  64
#define HH  512
#define G4H (4*HH)

typedef __attribute__((ext_vector_type(16))) __bf16 v16bf;
typedef __attribute__((ext_vector_type(8)))  __bf16 v8bf;
typedef __attribute__((ext_vector_type(8)))  float  v8f;

union AFrag { v16bf v; v8bf h[2]; };

#define WMMA_BF16(a, b, c) \
  __builtin_amdgcn_wmma_f32_16x16x32_bf16(false, (a), false, (b), (short)0, (c), false, false)

// ---------------------------------------------------------------------------
// Pack a row-major fp32 KxN matrix into per-lane WMMA B-fragments (bf16).
// Layout: frag[nt][kt][lane][16 halves], nt = N/16 tiles, kt = K/32 tiles.
// Lane n<16 holds column (nt*16+n), rows kt*32 + 0..7,16..23; lane n+16 rows 8..15,24..31.
// ---------------------------------------------------------------------------
__global__ void lstm_pack_b_kernel(const float* __restrict__ W,
                                   __bf16* __restrict__ out, int K, int N) {
  int idx = blockIdx.x * blockDim.x + threadIdx.x;
  int KT = K >> 5, NT = N >> 4;
  if (idx >= NT * KT * 32) return;
  int lane = idx & 31;
  int kt = (idx >> 5) % KT;
  int nt = (idx >> 5) / KT;
  int n  = nt * 16 + (lane & 15);
  int k0 = kt * 32 + ((lane & 16) ? 8 : 0);
  __bf16* dst = out + (size_t)idx * 16;
#pragma unroll
  for (int j = 0; j < 16; ++j)
    dst[j] = (__bf16)W[(size_t)(k0 + j + ((j >= 8) ? 8 : 0)) * N + n];
}

__global__ void lstm_cvt_bf16_kernel(const float* __restrict__ in,
                                     __bf16* __restrict__ out, int n) {
  int i = blockIdx.x * blockDim.x + threadIdx.x;
  if (i < n) out[i] = (__bf16)in[i];
}

__global__ void lstm_zero_state_kernel(float* __restrict__ c,
                                       __bf16* __restrict__ h0,
                                       __bf16* __restrict__ h1, int n) {
  int i = blockIdx.x * blockDim.x + threadIdx.x;
  if (i < n) { c[i] = 0.f; h0[i] = (__bf16)0.f; h1[i] = (__bf16)0.f; }
}

// ---------------------------------------------------------------------------
// Big parallel GEMM: out_bf16[M x N] = A_bf16[M x K] @ Bpacked + bias.
// Each wave: one 16-row M tile x four consecutive 16-col N tiles (A reuse x4).
// WMMA bf16 16x16x32, f32 accumulate. Prefetches are unconditional
// (speculative on gfx1250; OOB prefetch is silently dropped) to keep the
// inner loop branch-free.
// ---------------------------------------------------------------------------
__global__ void lstm_gemm_bias_kernel(const __bf16* __restrict__ A,
                                      const __bf16* __restrict__ Bp,
                                      const float* __restrict__ bias,
                                      __bf16* __restrict__ out,
                                      int M, int N, int K) {
  int lane = threadIdx.x & 31;
  int wid  = (blockIdx.x * blockDim.x + threadIdx.x) >> 5;
  int NG = N >> 6;                       // groups of 4 n-tiles
  int mt = wid / NG, ng = wid % NG;
  if (mt >= (M >> 4)) return;
  int KT = K >> 5;
  int laneM = lane & 15, hi = (lane >> 4) & 1;
  const __bf16* arow = A + (size_t)(mt * 16 + laneM) * K + hi * 8;
  v8f acc0 = {}, acc1 = {}, acc2 = {}, acc3 = {};
  const __bf16* bbase = Bp + ((size_t)(ng * 4) * KT * 32 + lane) * 16;
  const size_t bstride = (size_t)KT * 32 * 16;  // elements between n-tiles
  for (int kt = 0; kt < KT; ++kt) {
    AFrag a;
    a.h[0] = *(const v8bf*)(arow + kt * 32);
    a.h[1] = *(const v8bf*)(arow + kt * 32 + 16);
    const __bf16* bp = bbase + (size_t)kt * 32 * 16;
    __builtin_prefetch(bp + 32 * 16, 0, 1);                // next k-step, tile 0
    __builtin_prefetch(bp + bstride + 32 * 16, 0, 1);      // next k-step, tile 1
    v16bf b0 = *(const v16bf*)(bp);
    v16bf b1 = *(const v16bf*)(bp + bstride);
    v16bf b2 = *(const v16bf*)(bp + 2 * bstride);
    v16bf b3 = *(const v16bf*)(bp + 3 * bstride);
    acc0 = WMMA_BF16(a.v, b0, acc0);
    acc1 = WMMA_BF16(a.v, b1, acc1);
    acc2 = WMMA_BF16(a.v, b2, acc2);
    acc3 = WMMA_BF16(a.v, b3, acc3);
  }
  // C/D layout: VGPR r, lanes 0-15 -> (M=r, N=lane); lanes 16-31 -> (M=8+r).
  v8f accs[4] = {acc0, acc1, acc2, acc3};
#pragma unroll
  for (int j = 0; j < 4; ++j) {
    int nb = (ng * 4 + j) * 16 + laneM;
    float bv = bias[nb];
#pragma unroll
    for (int r = 0; r < 8; ++r) {
      int mrow = mt * 16 + hi * 8 + r;
      out[(size_t)mrow * N + nb] = (__bf16)(accs[j][r] + bv);
    }
  }
}

// ---------------------------------------------------------------------------
// One LSTM timestep (recurrent-only matmul, input projection precomputed):
//   z = xW[:, t, :] + h_{t-1} @ Wh ;  gates ; c,h update.
// Each wave: one 16-batch M tile x one 16-col tile of H, computing all 4 gate
// tiles (N, N+H, N+2H, N+3H) so the gate nonlinearity is local to the wave.
// 128 waves total (latency-bound sequential step; weights resident in L2).
// ---------------------------------------------------------------------------
__global__ void lstm_step_kernel(const __bf16* __restrict__ xW,   // (B,T,4H), bias folded in
                                 const __bf16* __restrict__ Whp,  // packed (4H/16,H/32,32,16)
                                 float* __restrict__ c,           // (B,H) f32 state
                                 const __bf16* __restrict__ hin,  // (B,H) h_{t-1}
                                 __bf16* __restrict__ hout,       // (B,H) h_t
                                 __bf16* __restrict__ hseq,       // (B,T,H) or null
                                 float* __restrict__ hf32,        // (B,H) or null
                                 int t) {
  int lane = threadIdx.x & 31;
  int wid = blockIdx.x * (blockDim.x >> 5) + (threadIdx.x >> 5);  // 0..127
  int mt = wid >> 5;   // 0..3   (B/16)
  int nt = wid & 31;   // 0..31  (H/16)
  int laneM = lane & 15, hi = (lane >> 4) & 1;
  const int KT = HH >> 5;  // 16
  const __bf16* arow = hin + (size_t)(mt * 16 + laneM) * HH + hi * 8;
  // Gate g's B tile lives at n-tile (g*H/16 + nt): constant stride per gate.
  const size_t gstride = (size_t)(HH >> 4) * KT * 32 * 16;  // elements per gate block
  const __bf16* bbase = Whp + ((size_t)nt * KT * 32 + lane) * 16;
  v8f acc0 = {}, acc1 = {}, acc2 = {}, acc3 = {};
  for (int kt = 0; kt < KT; ++kt) {
    AFrag a;
    a.h[0] = *(const v8bf*)(arow + kt * 32);
    a.h[1] = *(const v8bf*)(arow + kt * 32 + 16);
    const __bf16* bp = bbase + (size_t)kt * 32 * 16;
    __builtin_prefetch(bp + 32 * 16, 0, 1);                // speculative; branch-free
    __builtin_prefetch(bp + gstride + 32 * 16, 0, 1);
    v16bf b0 = *(const v16bf*)(bp);
    v16bf b1 = *(const v16bf*)(bp + gstride);
    v16bf b2 = *(const v16bf*)(bp + 2 * gstride);
    v16bf b3 = *(const v16bf*)(bp + 3 * gstride);
    acc0 = WMMA_BF16(a.v, b0, acc0);
    acc1 = WMMA_BF16(a.v, b1, acc1);
    acc2 = WMMA_BF16(a.v, b2, acc2);
    acc3 = WMMA_BF16(a.v, b3, acc3);
  }
  int n = nt * 16 + laneM;
#pragma unroll
  for (int r = 0; r < 8; ++r) {
    int mr = mt * 16 + hi * 8 + r;
    size_t xoff = ((size_t)mr * TT + t) * G4H;
    float zi = acc0[r] + (float)xW[xoff + n];
    float zf = acc1[r] + (float)xW[xoff + n + HH];
    float zg = acc2[r] + (float)xW[xoff + n + 2 * HH];
    float zo = acc3[r] + (float)xW[xoff + n + 3 * HH];
    float iv = 1.0f / (1.0f + __expf(-zi));
    float fv = 1.0f / (1.0f + __expf(-zf));
    float gv = tanhf(zg);
    float ov = 1.0f / (1.0f + __expf(-zo));
    size_t so = (size_t)mr * HH + n;
    float cv = fv * c[so] + iv * gv;
    c[so] = cv;
    float hv = ov * tanhf(cv);
    hout[so] = (__bf16)hv;
    if (hseq) hseq[((size_t)mr * TT + t) * HH + n] = (__bf16)hv;
    if (hf32) hf32[so] = hv;
  }
}

// ---------------------------------------------------------------------------
// Dense head: out = relu(relu(h2 @ Wd1 + bd1) @ Wd2 + bd2). Tiny (~8 MFLOP).
// One block per batch row; LDS tree reduction for the final dot.
// ---------------------------------------------------------------------------
__global__ void lstm_dense_head_kernel(const float* __restrict__ h2,
                                       const float* __restrict__ Wd1,
                                       const float* __restrict__ bd1,
                                       const float* __restrict__ Wd2,
                                       const float* __restrict__ bd2,
                                       float* __restrict__ out) {
  __shared__ float red[HH / 2];
  int m = blockIdx.x, j = threadIdx.x;  // 256 threads
  float s = 0.f;
  for (int k = 0; k < HH; ++k)
    s += h2[(size_t)m * HH + k] * Wd1[(size_t)k * (HH / 2) + j];
  float hd = fmaxf(s + bd1[j], 0.f);
  red[j] = hd * Wd2[j];
  __syncthreads();
  for (int s2 = HH / 4; s2 > 0; s2 >>= 1) {
    if (j < s2) red[j] += red[j + s2];
    __syncthreads();
  }
  if (j == 0) out[m] = fmaxf(red[0] + bd2[0], 0.f);
}

extern "C" void kernel_launch(void* const* d_in, const int* in_sizes, int n_in,
                              void* d_out, int out_size, void* d_ws, size_t ws_size,
                              hipStream_t stream) {
  const float* x   = (const float*)d_in[0];
  // d_in[1] = training flag (inference path only)
  const float* Wi1 = (const float*)d_in[2];
  const float* Wh1 = (const float*)d_in[3];
  const float* b1  = (const float*)d_in[4];
  const float* Wi2 = (const float*)d_in[5];
  const float* Wh2 = (const float*)d_in[6];
  const float* b2  = (const float*)d_in[7];
  const float* Wd1 = (const float*)d_in[8];
  const float* bd1 = (const float*)d_in[9];
  const float* Wd2 = (const float*)d_in[10];
  const float* bd2 = (const float*)d_in[11];
  float* out = (float*)d_out;

  // Workspace carve-up (all offsets 256B aligned). Total ~171 MB.
  char* ws = (char*)d_ws;
  size_t off = 0;
  auto alloc = [&](size_t bytes) -> char* {
    char* p = ws + off;
    off += (bytes + 255) & ~(size_t)255;
    return p;
  };
  __bf16* Wi1p  = (__bf16*)alloc((size_t)(G4H / 16) * (FF / 32) * 32 * 16 * 2);
  __bf16* Wh1p  = (__bf16*)alloc((size_t)(G4H / 16) * (HH / 32) * 32 * 16 * 2);
  __bf16* Wi2p  = (__bf16*)alloc((size_t)(G4H / 16) * (HH / 32) * 32 * 16 * 2);
  __bf16* Wh2p  = (__bf16*)alloc((size_t)(G4H / 16) * (HH / 32) * 32 * 16 * 2);
  __bf16* xbf   = (__bf16*)alloc((size_t)BSZ * TT * FF * 2);
  __bf16* h1seq = (__bf16*)alloc((size_t)BSZ * TT * HH * 2);
  __bf16* xW    = (__bf16*)alloc((size_t)BSZ * TT * G4H * 2);  // reused for both layers
  float*  cst   = (float*)alloc((size_t)BSZ * HH * 4);
  __bf16* hbuf0 = (__bf16*)alloc((size_t)BSZ * HH * 2);
  __bf16* hbuf1 = (__bf16*)alloc((size_t)BSZ * HH * 2);
  float*  hf32  = (float*)alloc((size_t)BSZ * HH * 4);

  const int thr = 256;

  // Pack all weight matrices into WMMA B-fragment layout (one-time).
  lstm_pack_b_kernel<<<((G4H / 16) * (FF / 32) * 32 + thr - 1) / thr, thr, 0, stream>>>(Wi1, Wi1p, FF, G4H);
  lstm_pack_b_kernel<<<((G4H / 16) * (HH / 32) * 32 + thr - 1) / thr, thr, 0, stream>>>(Wh1, Wh1p, HH, G4H);
  lstm_pack_b_kernel<<<((G4H / 16) * (HH / 32) * 32 + thr - 1) / thr, thr, 0, stream>>>(Wi2, Wi2p, HH, G4H);
  lstm_pack_b_kernel<<<((G4H / 16) * (HH / 32) * 32 + thr - 1) / thr, thr, 0, stream>>>(Wh2, Wh2p, HH, G4H);

  // Convert input sequence to bf16 (A matrix of GEMM1).
  int nx = BSZ * TT * FF;
  lstm_cvt_bf16_kernel<<<(nx + thr - 1) / thr, thr, 0, stream>>>(x, xbf, nx);

  // Phase 1: xW1 = x @ Wi1 + b1  (M=32768, N=2048, K=64) — parallel WMMA GEMM.
  const int M = BSZ * TT;
  const int gemm_blocks = ((M / 16) * (G4H / 64)) / 8;  // 8 waves / 256-thread block
  lstm_gemm_bias_kernel<<<gemm_blocks, 256, 0, stream>>>(xbf, Wi1p, b1, xW, M, G4H, FF);

  // Phase 2: layer-1 recurrent scan (512 sequential launches; h ping-pong).
  int ns = BSZ * HH;
  lstm_zero_state_kernel<<<(ns + thr - 1) / thr, thr, 0, stream>>>(cst, hbuf0, hbuf1, ns);
  for (int t = 0; t < TT; ++t) {
    __bf16* hin  = (t & 1) ? hbuf1 : hbuf0;
    __bf16* hout = (t & 1) ? hbuf0 : hbuf1;
    lstm_step_kernel<<<32, 128, 0, stream>>>(xW, Wh1p, cst, hin, hout, h1seq, nullptr, t);
  }

  // Phase 3: xW2 = h1seq @ Wi2 + b2  (M=32768, N=2048, K=512).
  lstm_gemm_bias_kernel<<<gemm_blocks, 256, 0, stream>>>(h1seq, Wi2p, b2, xW, M, G4H, HH);

  // Phase 4: layer-2 recurrent scan; only final h needed (fp32 for the head).
  lstm_zero_state_kernel<<<(ns + thr - 1) / thr, thr, 0, stream>>>(cst, hbuf0, hbuf1, ns);
  for (int t = 0; t < TT; ++t) {
    __bf16* hin  = (t & 1) ? hbuf1 : hbuf0;
    __bf16* hout = (t & 1) ? hbuf0 : hbuf1;
    lstm_step_kernel<<<32, 128, 0, stream>>>(xW, Wh2p, cst, hin, hout, nullptr,
                                             (t == TT - 1) ? hf32 : nullptr, t);
  }

  // Phase 5: dense head.
  lstm_dense_head_kernel<<<BSZ, HH / 2, 0, stream>>>(hf32, Wd1, bd1, Wd2, bd2, out);
}
